// UnidirectionalCrossAttention_10711648436563
// MI455X (gfx1250) — compile-verified, matrix operand order checked by reading, not run on hardware
//
#include <hip/hip_runtime.h>
#include <stdint.h>

#define HEADS 12
#define HDIM 64
#define SEQ 4096
#define CIN 768
#define QKSCALE 0.125f   // 64^-0.5

typedef __attribute__((ext_vector_type(16))) __bf16   v16bf;
typedef __attribute__((ext_vector_type(8)))  float    v8f;
typedef __attribute__((ext_vector_type(16))) uint16_t v16u;
typedef __attribute__((ext_vector_type(8)))  uint16_t v8u;
typedef __attribute__((ext_vector_type(4)))  float    f4;
typedef __attribute__((ext_vector_type(4)))  uint32_t u32x4;
typedef __attribute__((ext_vector_type(8)))  int      i32x8;
typedef __attribute__((ext_vector_type(4)))  int      i32x4;

#if defined(__AMDGCN__) && __has_builtin(__builtin_amdgcn_tensor_load_to_lds)
#define USE_TDM 1
#else
#define USE_TDM 0
#endif

// round-to-nearest-even f32 -> bf16 (bit pattern in uint16_t)
static __device__ __forceinline__ uint16_t f2bf(float x) {
  uint32_t u = __builtin_bit_cast(uint32_t, x);
  u += 0x7FFFu + ((u >> 16) & 1u);
  return (uint16_t)(u >> 16);
}

// Load a 16x32 bf16 WMMA A/B fragment from a row-major [16][>=32] tile.
// ISA layout: lanes 0-15 hold row l, k {0..7,16..23}; lanes 16-31 hold row
// (l-16), k {8..15,24..31}.  Two 16B loads per lane.
static __device__ __forceinline__ v16u ld_frag(const uint16_t* base, int ld) {
  const int lane = threadIdx.x & 31;
  const uint16_t* p = base + (lane & 15) * ld + ((lane >> 4) << 3);
  v8u lo = *(const v8u*)p;
  v8u hi = *(const v8u*)(p + 16);
  return __builtin_shufflevector(lo, hi, 0, 1, 2, 3, 4, 5, 6, 7,
                                         8, 9, 10, 11, 12, 13, 14, 15);
}

static __device__ __forceinline__ v8f wmma_bf16(v16u a, v16u b, v8f c) {
  return __builtin_amdgcn_wmma_f32_16x16x32_bf16(
      false, __builtin_bit_cast(v16bf, a),
      false, __builtin_bit_cast(v16bf, b),
      (short)0, c, false, false);
}

// Out(Mr x Nout) = A(Mr x K, f32 row-major) * W(Nout x K, f32 row-major)^T
// Block: 256 thr (8 waves), 64x64 output tile, BK=64 bf16 LDS staging.
// mode 0: Nout=1536, split qv -> o0=Q[h][tok][d], o1=V[h][tok][d] (bf16)
// mode 1: Nout=768,  -> o0=K[h][tok][d] (bf16), scaled by QKSCALE
// mode 2: Nout=768,  -> of[tok][c] f32 + bias
__global__ __launch_bounds__(256) void gemm_nt(
    const float* __restrict__ A, const float* __restrict__ W, int K, int mode,
    float scale, uint16_t* __restrict__ o0, uint16_t* __restrict__ o1,
    float* __restrict__ of, const float* __restrict__ bias) {
  __shared__ uint16_t As[64][72];  // +8 pad: conflict-free b128 frag reads
  __shared__ uint16_t Ws[64][72];

  const int tid = threadIdx.x;
  const int m0 = blockIdx.y * 64;
  const int n0 = blockIdx.x * 64;
  const int w  = tid >> 5;
  const int wr = (w & 3) * 16;   // wave row strip
  const int wc = (w >> 2) * 32;  // wave col strip

  v8f acc[2];
  acc[0] = {};
  acc[1] = {};

  const int r = tid >> 2;         // 0..63
  const int c = (tid & 3) * 16;   // 0,16,32,48

  for (int k0 = 0; k0 < K; k0 += 64) {
    const f4* pa = (const f4*)(A + (size_t)(m0 + r) * K + k0 + c);
    const f4* pw = (const f4*)(W + (size_t)(n0 + r) * K + k0 + c);
    f4 av[4], wv4[4];
#pragma unroll
    for (int u = 0; u < 4; u++) { av[u] = pa[u]; wv4[u] = pw[u]; }
    if (k0 + 64 < K) {
      __builtin_prefetch((const void*)(pa + 16), 0, 3);
      __builtin_prefetch((const void*)(pw + 16), 0, 3);
    }
    __syncthreads();  // previous-iter fragment reads done
#pragma unroll
    for (int u = 0; u < 4; u++)
#pragma unroll
      for (int j = 0; j < 4; j++) {
        As[r][c + u * 4 + j] = f2bf(av[u][j]);
        Ws[r][c + u * 4 + j] = f2bf(wv4[u][j]);
      }
    __syncthreads();

    v16u a0  = ld_frag(&As[wr][0], 72);
    v16u a1  = ld_frag(&As[wr][0] + 32, 72);
    v16u b00 = ld_frag(&Ws[wc][0], 72);
    v16u b01 = ld_frag(&Ws[wc][0] + 32, 72);
    v16u b10 = ld_frag(&Ws[wc + 16][0], 72);
    v16u b11 = ld_frag(&Ws[wc + 16][0] + 32, 72);
    acc[0] = wmma_bf16(a0, b00, acc[0]);
    acc[0] = wmma_bf16(a1, b01, acc[0]);
    acc[1] = wmma_bf16(a0, b10, acc[1]);
    acc[1] = wmma_bf16(a1, b11, acc[1]);
  }

  const int lane = tid & 31;
  const int hl = lane >> 4;
  const int l  = lane & 15;
#pragma unroll
  for (int t = 0; t < 2; t++) {
#pragma unroll
    for (int i = 0; i < 8; i++) {
      int gr = m0 + wr + i + hl * 8;
      int gc = n0 + wc + t * 16 + l;
      float v = acc[t][i] * scale;
      if (mode == 0) {
        int sel = gc / CIN;
        int rem = gc - sel * CIN;
        int hh = rem >> 6, d = rem & 63;
        uint16_t* dst = sel ? o1 : o0;
        dst[(((size_t)hh * SEQ + gr) << 6) + d] = f2bf(v);
      } else if (mode == 1) {
        int hh = gc >> 6, d = gc & 63;
        o0[(((size_t)hh * SEQ + gr) << 6) + d] = f2bf(v);
      } else {
        of[(size_t)gr * CIN + gc] = v + bias[gc];
      }
    }
  }
}

// Flash attention. attn[h,n,m] = K[h,n,:] . Q[h,m,:] (K pre-scaled), softmax
// over m, Y[n, h*64+d] = sum_m P * V[h,m,d].
// Grid (SEQ/64, HEADS); block 128 (4 waves); wave = 16 target rows.
// Q source tiles are staged global->LDS by the Tensor Data Mover (with D#
// padding reproducing the [64][72] conflict-free layout); V goes through
// registers because it needs an element transpose.
__global__ __launch_bounds__(128) void attn_kernel(
    const uint16_t* __restrict__ Qb, const uint16_t* __restrict__ Kb,
    const uint16_t* __restrict__ Vb, float* __restrict__ Yb) {
  __shared__ uint16_t Qs[64][72];      // source tile [m][d]
  __shared__ uint16_t Vt[64][72];      // V transposed [d][m]
  __shared__ uint16_t Ps[4][16][72];   // per-wave P bounce [row][m]

  const int tid  = threadIdx.x;
  const int lane = tid & 31;
  const int wv   = tid >> 5;
  const int hl   = lane >> 4;
  const int l    = lane & 15;
  const int h    = blockIdx.y;
  const int n0   = blockIdx.x * 64;

  // A-side (target K rows) fragments: resident for the whole pass
  const uint16_t* kbase = Kb + (((size_t)h * SEQ + n0 + wv * 16) << 6);
  v16u kA0 = ld_frag(kbase, 64);
  v16u kA1 = ld_frag(kbase + 32, 64);

  float mi[8], li[8];
  v8f o[4];
#pragma unroll
  for (int i = 0; i < 8; i++) { mi[i] = -INFINITY; li[i] = 0.f; }
#pragma unroll
  for (int dt = 0; dt < 4; dt++) o[dt] = {};

  const int rr   = tid >> 1;        // 0..63 (source row within tile)
  const int dseg = (tid & 1) * 32;  // d segment

  for (int m0 = 0; m0 < SEQ; m0 += 64) {
    const size_t qoff = ((size_t)h * SEQ + m0) << 6;
    const uint16_t* vg = Vb + qoff + ((size_t)rr << 6) + dseg;
    v8u vvv[4];
#pragma unroll
    for (int u = 0; u < 4; u++) vvv[u] = ((const v8u*)vg)[u];
#if !USE_TDM
    const uint16_t* qg = Qb + qoff + ((size_t)rr << 6) + dseg;
    v8u q[4];
#pragma unroll
    for (int u = 0; u < 4; u++) q[u] = ((const v8u*)qg)[u];
#endif
    if (m0 + 64 < SEQ)
      __builtin_prefetch((const void*)(vg + (64 << 6)), 0, 3);
    __syncthreads();  // all waves done reading prev LDS tiles

#if USE_TDM
    if (wv == 0) {
      // Tensor DMA descriptor (ISA ch8): 2D bf16 tensor, tile 64x64,
      // LDS padding 4 DWORDs every 32 DWORDs -> [64][72] layout.
      uint64_t ga = (uint64_t)(uintptr_t)(Qb + qoff);
      uint32_t la = (uint32_t)(uintptr_t)(&Qs[0][0]);
      u32x4 g0 = {};
      g0[0] = 1u;                                   // count=1, user mode
      g0[1] = la;                                   // lds_addr
      g0[2] = (uint32_t)ga;                         // global_addr[31:0]
      g0[3] = ((uint32_t)(ga >> 32) & 0x01FFFFFFu)  // global_addr[56:32]
              | 0x80000000u;                        // type=2 ("image")
      i32x8 g1 = {};
      g1[0] = (1 << 16)      // data_size = 2 bytes
            | (1 << 20)      // pad_enable
            | (4 << 22)      // pad_interval: 32 DWORDs (one 64-elem row)
            | (3 << 25);     // pad_amount: 4 DWORDs (8 bf16)
      g1[1] = (64 << 16);            // tensor_dim0[15:0] = 64 (d)
      g1[2] = (int)(4096u << 16);    // tensor_dim1[15:0] = 4096 (m)
      g1[3] = (64 << 16);            // tile_dim0 = 64
      g1[4] = 64;                    // tile_dim1 = 64, tile_dim2 = 0
      g1[5] = 64;                    // tensor_dim0_stride = 64
      i32x4 gz4 = {};
      i32x8 gz8 = {};
      // amdgpu-toolchain (clang-23) 6-arg form:
      // (g0, g1, group2, group3, extra, cpol)
      __builtin_amdgcn_tensor_load_to_lds(g0, g1, gz4, gz4, gz8, 0);
    }
#else
#pragma unroll
    for (int u = 0; u < 4; u++) ((v8u*)&Qs[rr][dseg])[u] = q[u];
#endif
    // V transpose scatter (registers -> LDS)
#pragma unroll
    for (int u = 0; u < 4; u++)
#pragma unroll
      for (int j = 0; j < 8; j++) Vt[dseg + u * 8 + j][rr] = vvv[u][j];
#if USE_TDM
    if (wv == 0) __builtin_amdgcn_s_wait_tensorcnt(0);
#endif
    __syncthreads();

    // S = K_strip (16xD) x Q_tile^T -> 16x64, in 4 WMMA accs
    v8f s[4];
#pragma unroll
    for (int mc = 0; mc < 4; mc++) {
      v8f z = {};
      z = wmma_bf16(kA0, ld_frag(&Qs[mc * 16][0], 72), z);
      z = wmma_bf16(kA1, ld_frag(&Qs[mc * 16][0] + 32, 72), z);
      s[mc] = z;
    }

    // online softmax (row stats replicated in each 16-lane half)
    float al[8];
#pragma unroll
    for (int i = 0; i < 8; i++) {
      float t = fmaxf(fmaxf(s[0][i], s[1][i]), fmaxf(s[2][i], s[3][i]));
      t = fmaxf(t, __shfl_xor(t, 1, 16));
      t = fmaxf(t, __shfl_xor(t, 2, 16));
      t = fmaxf(t, __shfl_xor(t, 4, 16));
      t = fmaxf(t, __shfl_xor(t, 8, 16));
      float mn = fmaxf(mi[i], t);
      float a  = __expf(mi[i] - mn);
      mi[i] = mn;
      float sum = 0.f;
#pragma unroll
      for (int mc = 0; mc < 4; mc++) {
        float p = __expf(s[mc][i] - mn);
        s[mc][i] = p;
        sum += p;
      }
      sum += __shfl_xor(sum, 1, 16);
      sum += __shfl_xor(sum, 2, 16);
      sum += __shfl_xor(sum, 4, 16);
      sum += __shfl_xor(sum, 8, 16);
      li[i] = li[i] * a + sum;
      al[i] = a;
    }
#pragma unroll
    for (int dt = 0; dt < 4; dt++)
#pragma unroll
      for (int i = 0; i < 8; i++) o[dt][i] *= al[i];

    // C-layout -> A-layout via per-wave LDS bounce (bf16)
#pragma unroll
    for (int mc = 0; mc < 4; mc++)
#pragma unroll
      for (int i = 0; i < 8; i++)
        Ps[wv][i + hl * 8][mc * 16 + l] = f2bf(s[mc][i]);

    // O += P (16x64) x V (64x64)
#pragma unroll
    for (int dt = 0; dt < 4; dt++) {
      v8f oo = o[dt];
      oo = wmma_bf16(ld_frag(&Ps[wv][0][0], 72),
                     ld_frag(&Vt[dt * 16][0], 72), oo);
      oo = wmma_bf16(ld_frag(&Ps[wv][0][0] + 32, 72),
                     ld_frag(&Vt[dt * 16][0] + 32, 72), oo);
      o[dt] = oo;
    }
  }

#pragma unroll
  for (int dt = 0; dt < 4; dt++)
#pragma unroll
    for (int i = 0; i < 8; i++) {
      int n   = n0 + wv * 16 + i + hl * 8;
      int col = h * 64 + dt * 16 + l;
      Yb[(size_t)n * CIN + col] = o[dt][i] / li[i];
    }
}

extern "C" void kernel_launch(void* const* d_in, const int* in_sizes, int n_in,
                              void* d_out, int out_size, void* d_ws,
                              size_t ws_size, hipStream_t stream) {
  (void)in_sizes; (void)n_in; (void)out_size; (void)ws_size;
  const float* x      = (const float*)d_in[0];
  const float* src    = (const float*)d_in[1];
  const float* W_qv   = (const float*)d_in[2];
  const float* W_k    = (const float*)d_in[3];
  const float* W_proj = (const float*)d_in[4];
  const float* b_proj = (const float*)d_in[5];
  float* out = (float*)d_out;

  uint8_t* ws = (uint8_t*)d_ws;
  const size_t hv = (size_t)HEADS * SEQ * HDIM * sizeof(uint16_t);  // 6 MB
  uint16_t* Qb = (uint16_t*)(ws);
  uint16_t* Vb = (uint16_t*)(ws + hv);
  uint16_t* Kb = (uint16_t*)(ws + 2 * hv);
  float*    Yb = (float*)(ws + 3 * hv);  // 12.6 MB f32

  // qv = src @ W_qv^T -> Q,V (bf16, head-major)
  gemm_nt<<<dim3(2 * CIN / 64, SEQ / 64), 256, 0, stream>>>(
      src, W_qv, CIN, 0, 1.0f, Qb, Vb, nullptr, nullptr);
  // k = (x @ W_k^T) * scale -> K (bf16, head-major)
  gemm_nt<<<dim3(CIN / 64, SEQ / 64), 256, 0, stream>>>(
      x, W_k, CIN, 1, QKSCALE, Kb, nullptr, nullptr, nullptr);
  // flash attention -> Yb (f32, [n][h*64+d])
  attn_kernel<<<dim3(SEQ / 64, HEADS), 128, 0, stream>>>(Qb, Kb, Vb, Yb);
  // out = Yb @ W_proj^T + b_proj
  gemm_nt<<<dim3(CIN / 64, SEQ / 64), 256, 0, stream>>>(
      Yb, W_proj, CIN, 2, 1.0f, nullptr, nullptr, out, b_proj);
}
